// BSBRAttention_16535624090009
// MI455X (gfx1250) — matrix operand order repaired and checked by reading, not compiled, not run on hardware
//
#include <hip/hip_runtime.h>
#include <hip/hip_bf16.h>

// Problem constants (match reference)
constexpr int Bc = 2, Sc = 4096, Dc = 1024, Hc = 16, CSc = 128;
constexpr int HDc = Dc / Hc;     // 64
constexpr int Cc  = Sc / CSc;    // 32
constexpr float SCALE = 0.125f;  // 1/sqrt(64)
constexpr float NEGF  = -1e9f;

typedef __attribute__((ext_vector_type(16))) __bf16 v16bf;
typedef __attribute__((ext_vector_type(8)))  float  v8f;

union ABFrag { v16bf v; unsigned short u[16]; };

__device__ inline unsigned short f2bf(float f) {
    union { float f; unsigned u; } x; x.f = f;
    unsigned r = x.u + 0x7FFFu + ((x.u >> 16) & 1u);   // RNE
    return (unsigned short)(r >> 16);
}
__device__ inline float bf2f(unsigned short b) {
    union { unsigned u; float f; } x; x.u = ((unsigned)b) << 16; return x.f;
}

__device__ inline v8f wmma_bf16(v16bf a, v16bf b, v8f c) {
    // D(f32,16x16) = A(bf16,16x32) * B(bf16,32x16) + C
    return __builtin_amdgcn_wmma_f32_16x16x32_bf16(false, a, false, b, (short)0, c,
                                                   false, false);
}

// ---------------------------------------------------------------------------
// Elementwise f32 -> bf16 convert (grid-stride)
// ---------------------------------------------------------------------------
__global__ void cvt_f32_bf16(const float* __restrict__ src,
                             unsigned short* __restrict__ dst, int n) {
    for (int i = blockIdx.x * blockDim.x + threadIdx.x; i < n;
         i += gridDim.x * blockDim.x)
        dst[i] = f2bf(src[i]);
}

// Gather chunk representatives (last token of each chunk), bf16 [B*C, D]
__global__ void gather_repr(const unsigned short* __restrict__ hsb,
                            unsigned short* __restrict__ crb) {
    int i = blockIdx.x * blockDim.x + threadIdx.x;
    if (i >= Bc * Cc * Dc) return;
    int d = i & (Dc - 1);
    int c = (i >> 10) & (Cc - 1);
    int b = i >> 15;
    crb[i] = hsb[((size_t)(b * Sc + c * CSc + CSc - 1)) * Dc + d];
}

// ---------------------------------------------------------------------------
// Generic WMMA GEMM: Out[M,N] = X[M,K] * W[N,K]^T + bias[N]
// X, W bf16 row-major.  Out: bf16 (out_bf) or f32 (out_f, if non-null).
// Block: 256 threads = 8 waves arranged 2(M) x 4(N); wave tile 64x32
// (4 A-frags x 2 B-frags -> 8 WMMA per K-step: 2x arithmetic intensity vs 32x32).
// Requires N % 128 == 0; M handled by per-wave early-out (multiple of 64).
// ---------------------------------------------------------------------------
__global__ void gemm_xwt_wmma(const unsigned short* __restrict__ X, int ldx,
                              const unsigned short* __restrict__ W, int ldw,
                              const float* __restrict__ bias,
                              int M, int N,
                              unsigned short* __restrict__ out_bf,
                              float* __restrict__ out_f) {
    const int K = ldx;
    const int lane = threadIdx.x & 31;
    const int wave = threadIdx.x >> 5;
    const int m0 = blockIdx.x * 128 + (wave >> 2) * 64;
    const int n0 = blockIdx.y * 128 + (wave & 3) * 32;
    if (m0 >= M) return;

    const int l15 = lane & 15;
    const int hi8 = ((lane >> 4) & 1) * 8;    // accumulator row base (rows 8h..8h+7)
    const int hi16 = ((lane >> 4) & 1) * 16;  // B-fragment K base

    const float bias0 = bias ? bias[n0 + l15] : 0.f;
    const float bias1 = bias ? bias[n0 + 16 + l15] : 0.f;
    v8f acc[4][2];
#pragma unroll
    for (int mi = 0; mi < 4; mi++)
#pragma unroll
        for (int j = 0; j < 8; j++) { acc[mi][0][j] = bias0; acc[mi][1][j] = bias1; }

    const unsigned short* pa[4];
#pragma unroll
    for (int mi = 0; mi < 4; mi++)
        pa[mi] = X + (size_t)(m0 + mi * 16 + l15) * ldx;
    const unsigned short* pb0 = W + (size_t)(n0 + l15) * ldw;
    const unsigned short* pb1 = W + (size_t)(n0 + 16 + l15) * ldw;

    for (int k0 = 0; k0 < K; k0 += 32) {
        __builtin_prefetch(pa[0] + k0 + 32, 0, 0);  // global_prefetch_b8
        __builtin_prefetch(pb0 + k0 + 32, 0, 0);
        ABFrag a[4], b0, b1;
#pragma unroll
        for (int mi = 0; mi < 4; mi++)
#pragma unroll
            for (int j = 0; j < 16; j++) {
                a[mi].u[j]     = pa[mi][k0 + hi16 + j];
            }
#pragma unroll
        for (int j = 0; j < 16; j++) {
            b0.u[j] = pb0[k0 + hi16 + j];
            b1.u[j] = pb1[k0 + hi16 + j];
        }
#pragma unroll
        for (int mi = 0; mi < 4; mi++) {
            acc[mi][0] = wmma_bf16(a[mi].v, b0.v, acc[mi][0]);
            acc[mi][1] = wmma_bf16(a[mi].v, b1.v, acc[mi][1]);
        }
    }

    const int col0 = n0 + l15, col1 = n0 + 16 + l15;
#pragma unroll
    for (int mi = 0; mi < 4; mi++) {
#pragma unroll
        for (int j = 0; j < 8; j++) {
            int r = m0 + mi * 16 + hi8 + j;
            if (out_f) {
                out_f[(size_t)r * N + col0] = acc[mi][0][j];
                out_f[(size_t)r * N + col1] = acc[mi][1][j];
            } else {
                out_bf[(size_t)r * N + col0] = f2bf(acc[mi][0][j]);
                out_bf[(size_t)r * N + col1] = f2bf(acc[mi][1][j]);
            }
        }
    }
}

// ---------------------------------------------------------------------------
// F = K^T V per chunk: block = one (b,h,c); F[64,64] = sum_s K[s,d]*V[s,e]
// Stage K^T and V^T in LDS so all WMMA fragment reads are contiguous.
// ---------------------------------------------------------------------------
__global__ void chunk_f_kernel(const unsigned short* __restrict__ Kb,
                               const unsigned short* __restrict__ Vb,
                               unsigned short* __restrict__ Fb) {
    const int idx = blockIdx.x;                 // b*H*C + h*C + c
    const int c = idx & (Cc - 1);
    const int h = (idx / Cc) & (Hc - 1);
    const int b = idx / (Cc * Hc);
    extern __shared__ char smem[];
    unsigned short* Kt = (unsigned short*)smem;   // [64][128]
    unsigned short* Vt = Kt + 64 * 128;           // [64][128]
    const size_t base = ((size_t)b * Sc + (size_t)c * CSc) * Dc + h * HDc;

    for (int i = threadIdx.x; i < CSc * HDc; i += blockDim.x) {
        int s = i >> 6, d = i & 63;
        unsigned short kv = Kb[base + (size_t)s * Dc + d];
        unsigned short vv = Vb[base + (size_t)s * Dc + d];
        Kt[d * 128 + s] = kv;
        Vt[d * 128 + s] = vv;
    }
    __syncthreads();

    const int lane = threadIdx.x & 31, wave = threadIdx.x >> 5;
    const int l15 = lane & 15, hi8 = ((lane >> 4) & 1) * 8, hi16 = ((lane >> 4) & 1) * 16;
    const size_t fbase = (size_t)idx * 64 * 64;

    for (int t = wave; t < 16; t += 8) {        // 4x4 tiles of 16x16
        int mi = t >> 2, ni = t & 3;
        v8f acc;
#pragma unroll
        for (int j = 0; j < 8; j++) acc[j] = 0.f;
        for (int k0 = 0; k0 < 128; k0 += 32) {
            ABFrag a, bb;
            const unsigned short* pa = Kt + (mi * 16 + l15) * 128 + k0;
#pragma unroll
            for (int j = 0; j < 16; j++) { a.u[j] = pa[hi16 + j]; }
            const unsigned short* pb = Vt + (ni * 16 + l15) * 128 + k0 + hi16;
#pragma unroll
            for (int j = 0; j < 16; j++) bb.u[j] = pb[j];
            acc = wmma_bf16(a.v, bb.v, acc);
        }
        int e = ni * 16 + l15;
#pragma unroll
        for (int j = 0; j < 8; j++) {
            int d = mi * 16 + hi8 + j;
            Fb[fbase + (size_t)d * 64 + e] = f2bf(acc[j]);
        }
    }
}

// ---------------------------------------------------------------------------
// Chunk-level retrieval per (b,h): scores r·h^T (triu mask, softmax), then
// retrieved = P(32x32) @ F(32x4096) via WMMA.
// ---------------------------------------------------------------------------
__global__ void chunk_retrieve_kernel(const unsigned short* __restrict__ Rb,
                                      const unsigned short* __restrict__ Hb,
                                      const unsigned short* __restrict__ Fb,
                                      unsigned short* __restrict__ Retb) {
    const int bh = blockIdx.x;                  // b*H + h
    const int h = bh & (Hc - 1);
    const int b = bh >> 4;
    extern __shared__ char smem[];
    float* sc = (float*)smem;                            // [32][32]
    unsigned short* P = (unsigned short*)(sc + 32 * 32); // [32][32] bf16

    for (int i = threadIdx.x; i < Cc * Cc; i += blockDim.x) {
        int cc = i >> 5, e = i & 31;
        float v;
        if (e < cc) v = NEGF;                  // triu: attend e >= cc
        else {
            const unsigned short* pr = Rb + ((size_t)(b * Cc + cc)) * Dc + h * HDc;
            const unsigned short* ph = Hb + ((size_t)(b * Cc + e)) * Dc + h * HDc;
            float acc = 0.f;
#pragma unroll 8
            for (int x = 0; x < HDc; x++) acc += bf2f(pr[x]) * bf2f(ph[x]);
            v = acc * SCALE;
        }
        sc[i] = v;
    }
    __syncthreads();
    if (threadIdx.x < Cc) {
        int cc = threadIdx.x;
        float mx = -1e30f;
        for (int e = 0; e < Cc; e++) mx = fmaxf(mx, sc[cc * 32 + e]);
        float sum = 0.f;
        for (int e = 0; e < Cc; e++) sum += __expf(sc[cc * 32 + e] - mx);
        float inv = 1.f / sum;
        for (int e = 0; e < Cc; e++) P[cc * 32 + e] = f2bf(__expf(sc[cc * 32 + e] - mx) * inv);
    }
    __syncthreads();

    const int lane = threadIdx.x & 31, wave = threadIdx.x >> 5;
    const int l15 = lane & 15, hi8 = ((lane >> 4) & 1) * 8, hi16 = ((lane >> 4) & 1) * 16;
    const unsigned short* F = Fb + (size_t)bh * Cc * 64 * 64;  // [32][4096]
    unsigned short* R = Retb + (size_t)bh * Cc * 64 * 64;

    ABFrag a0, a1;  // P rows 0..15 and 16..31, K = 32 (single step)
    {
        const unsigned short* p0 = P + l15 * 32;
        const unsigned short* p1 = P + (16 + l15) * 32;
#pragma unroll
        for (int j = 0; j < 16; j++) {
            a0.u[j] = p0[hi16 + j];
            a1.u[j] = p1[hi16 + j];
        }
    }
    for (int nt = wave; nt < 256; nt += 8) {
        int n = nt * 16 + l15;
        ABFrag bb;
        const unsigned short* pb = F + (size_t)hi16 * 4096 + n;
#pragma unroll
        for (int j = 0; j < 16; j++) bb.u[j] = pb[(size_t)j * 4096];
        v8f acc0, acc1;
#pragma unroll
        for (int j = 0; j < 8; j++) { acc0[j] = 0.f; acc1[j] = 0.f; }
        acc0 = wmma_bf16(a0.v, bb.v, acc0);
        acc1 = wmma_bf16(a1.v, bb.v, acc1);
#pragma unroll
        for (int j = 0; j < 8; j++) {
            R[(size_t)(hi8 + j) * 4096 + n]      = f2bf(acc0[j]);
            R[(size_t)(16 + hi8 + j) * 4096 + n] = f2bf(acc1[j]);
        }
    }
}

// ---------------------------------------------------------------------------
// Main per-chunk attention: block = one (b,h,c).
//   S = Q K^T * scale, triu mask, softmax -> P (bf16)
//   out = P @ V + Q @ retrieved          (fused into one accumulator)
// LDS: Q[128][64], K[128][64], Vt[64][128], Rt[64][64], Sc f32[128][128],
//      P bf16[128][128]  => 152 KB (two workgroups fit per 320 KB WGP).
// ---------------------------------------------------------------------------
__global__ void chunk_attn_kernel(const unsigned short* __restrict__ Qb,
                                  const unsigned short* __restrict__ Kb,
                                  const unsigned short* __restrict__ Vb,
                                  const unsigned short* __restrict__ Retb,
                                  unsigned short* __restrict__ Ob) {
    const int idx = blockIdx.x;
    const int c = idx & 31;
    const int h = (idx >> 5) & 15;
    const int b = idx >> 9;
    extern __shared__ char smem[];
    unsigned short* Qs = (unsigned short*)smem;            // [128][64]
    unsigned short* Ks = Qs + 128 * 64;                    // [128][64]
    unsigned short* Vt = Ks + 128 * 64;                    // [64][128]
    unsigned short* Rt = Vt + 64 * 128;                    // [64][64]
    float* Scs = (float*)(Rt + 64 * 64);                   // [128][128]
    unsigned short* P = (unsigned short*)(Scs + 128 * 128);// [128][128]

    const size_t base = ((size_t)b * Sc + (size_t)c * CSc) * Dc + h * HDc;
    for (int i = threadIdx.x; i < CSc * HDc; i += blockDim.x) {
        int s = i >> 6, d = i & 63;
        Qs[s * 64 + d] = Qb[base + (size_t)s * Dc + d];
        Ks[s * 64 + d] = Kb[base + (size_t)s * Dc + d];
        Vt[d * 128 + s] = Vb[base + (size_t)s * Dc + d];
    }
    const unsigned short* Rg = Retb + (size_t)(((b * Hc) + h) * Cc + c) * 64 * 64;
    for (int i = threadIdx.x; i < 64 * 64; i += blockDim.x) {
        int x = i >> 6, y = i & 63;
        Rt[y * 64 + x] = Rg[i];
    }
    __syncthreads();

    const int lane = threadIdx.x & 31, wave = threadIdx.x >> 5;
    const int l15 = lane & 15, hi8 = ((lane >> 4) & 1) * 8, hi16 = ((lane >> 4) & 1) * 16;

    // ---- scores: 8x8 tiles of 16x16 over [128][128]
    for (int t = wave; t < 64; t += 8) {
        int mi = t >> 3, ni = t & 7;
        v8f acc;
#pragma unroll
        for (int j = 0; j < 8; j++) acc[j] = 0.f;
        for (int k0 = 0; k0 < 64; k0 += 32) {
            ABFrag a, bb;
            const unsigned short* pa = Qs + (mi * 16 + l15) * 64 + k0;
#pragma unroll
            for (int j = 0; j < 16; j++) { a.u[j] = pa[hi16 + j]; }
            const unsigned short* pb = Ks + (ni * 16 + l15) * 64 + k0 + hi16;
#pragma unroll
            for (int j = 0; j < 16; j++) bb.u[j] = pb[j];
            acc = wmma_bf16(a.v, bb.v, acc);
        }
        int n = ni * 16 + l15;
#pragma unroll
        for (int j = 0; j < 8; j++) {
            int m = mi * 16 + hi8 + j;
            float v = acc[j] * SCALE;
            if (n < m) v = NEGF;              // triu: attend t >= s
            Scs[m * 128 + n] = v;
        }
    }
    __syncthreads();

    // ---- row softmax -> P (bf16); each of threads 0..127 owns one row
    if (threadIdx.x < 128) {
        int m = threadIdx.x;
        float mx = -1e30f;
        for (int n = 0; n < 128; n++) mx = fmaxf(mx, Scs[m * 128 + n]);
        float sum = 0.f;
        for (int n = 0; n < 128; n++) { float e = __expf(Scs[m * 128 + n] - mx);
                                        Scs[m * 128 + n] = e; sum += e; }
        float inv = 1.f / sum;
        for (int n = 0; n < 128; n++) P[m * 128 + n] = f2bf(Scs[m * 128 + n] * inv);
    }
    __syncthreads();

    // ---- out = P@V + Q@Rt : 8(M) x 4(N) tiles of 16x16
    for (int t = wave; t < 32; t += 8) {
        int mi = t >> 2, ni = t & 3;
        v8f acc;
#pragma unroll
        for (int j = 0; j < 8; j++) acc[j] = 0.f;
        for (int k0 = 0; k0 < 128; k0 += 32) {          // P @ V
            ABFrag a, bb;
            const unsigned short* pa = P + (mi * 16 + l15) * 128 + k0;
#pragma unroll
            for (int j = 0; j < 16; j++) { a.u[j] = pa[hi16 + j]; }
            const unsigned short* pb = Vt + (ni * 16 + l15) * 128 + k0 + hi16;
#pragma unroll
            for (int j = 0; j < 16; j++) bb.u[j] = pb[j];
            acc = wmma_bf16(a.v, bb.v, acc);
        }
        for (int k0 = 0; k0 < 64; k0 += 32) {           // Q @ retrieved
            ABFrag a, bb;
            const unsigned short* pa = Qs + (mi * 16 + l15) * 64 + k0;
#pragma unroll
            for (int j = 0; j < 16; j++) { a.u[j] = pa[hi16 + j]; }
            const unsigned short* pb = Rt + (ni * 16 + l15) * 64 + k0 + hi16;
#pragma unroll
            for (int j = 0; j < 16; j++) bb.u[j] = pb[j];
            acc = wmma_bf16(a.v, bb.v, acc);
        }
        int d = ni * 16 + l15;
#pragma unroll
        for (int j = 0; j < 8; j++) {
            int s = mi * 16 + hi8 + j;
            Ob[base + (size_t)s * Dc + d] = f2bf(acc[j]);
        }
    }
}

// ---------------------------------------------------------------------------
extern "C" void kernel_launch(void* const* d_in, const int* in_sizes, int n_in,
                              void* d_out, int out_size, void* d_ws, size_t ws_size,
                              hipStream_t stream) {
    (void)in_sizes; (void)n_in; (void)out_size; (void)ws_size;
    const float* hs = (const float*)d_in[0];
    const float* Wq = (const float*)d_in[1];  const float* bq = (const float*)d_in[2];
    const float* Wk = (const float*)d_in[3];  const float* bk = (const float*)d_in[4];
    const float* Wv = (const float*)d_in[5];  const float* bv = (const float*)d_in[6];
    const float* Wo = (const float*)d_in[7];  const float* bo = (const float*)d_in[8];
    const float* Wr = (const float*)d_in[9];  const float* br = (const float*)d_in[10];
    const float* Wh = (const float*)d_in[11]; const float* bh = (const float*)d_in[12];

    const size_t nBSD = (size_t)Bc * Sc * Dc;   // 8.39M
    const size_t nW   = (size_t)Dc * Dc;        // 1.05M
    const size_t nBCD = (size_t)Bc * Cc * Dc;   // 65536
    const size_t nF   = (size_t)Bc * Hc * Cc * HDc * HDc; // 4.19M

    char* ws = (char*)d_ws;
    size_t off = 0;
    auto alloc = [&](size_t bytes) -> void* {
        void* p = ws + off; off += (bytes + 255) & ~(size_t)255; return p;
    };
    unsigned short* hsb  = (unsigned short*)alloc(nBSD * 2);
    unsigned short* wqb  = (unsigned short*)alloc(nW * 2);
    unsigned short* wkb  = (unsigned short*)alloc(nW * 2);
    unsigned short* wvb  = (unsigned short*)alloc(nW * 2);
    unsigned short* wob  = (unsigned short*)alloc(nW * 2);
    unsigned short* wrb  = (unsigned short*)alloc(nW * 2);
    unsigned short* whb  = (unsigned short*)alloc(nW * 2);
    unsigned short* qb   = (unsigned short*)alloc(nBSD * 2);
    unsigned short* kb   = (unsigned short*)alloc(nBSD * 2);
    unsigned short* vb   = (unsigned short*)alloc(nBSD * 2);
    unsigned short* attb = (unsigned short*)alloc(nBSD * 2);
    unsigned short* crb  = (unsigned short*)alloc(nBCD * 2);
    unsigned short* rb   = (unsigned short*)alloc(nBCD * 2);
    unsigned short* hbuf = (unsigned short*)alloc(nBCD * 2);
    unsigned short* fb   = (unsigned short*)alloc(nF * 2);
    unsigned short* retb = (unsigned short*)alloc(nF * 2);

    const int T = 256;
    // 1) convert to bf16
    cvt_f32_bf16<<<4096, T, 0, stream>>>(hs, hsb, (int)nBSD);
    cvt_f32_bf16<<<2048, T, 0, stream>>>(Wq, wqb, (int)nW);
    cvt_f32_bf16<<<2048, T, 0, stream>>>(Wk, wkb, (int)nW);
    cvt_f32_bf16<<<2048, T, 0, stream>>>(Wv, wvb, (int)nW);
    cvt_f32_bf16<<<2048, T, 0, stream>>>(Wo, wob, (int)nW);
    cvt_f32_bf16<<<2048, T, 0, stream>>>(Wr, wrb, (int)nW);
    cvt_f32_bf16<<<2048, T, 0, stream>>>(Wh, whb, (int)nW);
    gather_repr<<<(int)(nBCD / T), T, 0, stream>>>(hsb, crb);

    // 2) projections (WMMA GEMMs): block tile 128x128, wave tile 64x32
    dim3 gBig((Bc * Sc + 127) / 128, Dc / 128);
    gemm_xwt_wmma<<<gBig, T, 0, stream>>>(hsb, Dc, wqb, Dc, bq, Bc * Sc, Dc, qb, nullptr);
    gemm_xwt_wmma<<<gBig, T, 0, stream>>>(hsb, Dc, wkb, Dc, bk, Bc * Sc, Dc, kb, nullptr);
    gemm_xwt_wmma<<<gBig, T, 0, stream>>>(hsb, Dc, wvb, Dc, bv, Bc * Sc, Dc, vb, nullptr);
    dim3 gSm((Bc * Cc + 127) / 128, Dc / 128);
    gemm_xwt_wmma<<<gSm, T, 0, stream>>>(crb, Dc, wrb, Dc, br, Bc * Cc, Dc, rb, nullptr);
    gemm_xwt_wmma<<<gSm, T, 0, stream>>>(crb, Dc, whb, Dc, bh, Bc * Cc, Dc, hbuf, nullptr);

    // 3) F = K^T V per chunk
    chunk_f_kernel<<<Bc * Hc * Cc, T, 2 * 64 * 128 * 2, stream>>>(kb, vb, fb);

    // 4) chunk softmax + retrieved = P @ F
    chunk_retrieve_kernel<<<Bc * Hc, T, 32 * 32 * 4 + 32 * 32 * 2, stream>>>(rb, hbuf, fb, retb);

    // 5) local attention + long-term, fused
    size_t smem5 = (size_t)(128 * 64 * 2) * 2 + 64 * 128 * 2 + 64 * 64 * 2
                 + 128 * 128 * 4 + 128 * 128 * 2;   // 152 KB
    chunk_attn_kernel<<<Bc * Hc * Cc, T, smem5, stream>>>(qb, kb, vb, retb, attb);

    // 6) output projection -> f32 d_out
    gemm_xwt_wmma<<<gBig, T, 0, stream>>>(attb, Dc, wob, Dc, bo, Bc * Sc, Dc,
                                          nullptr, (float*)d_out);
}
